// GraspEvaluatorNetwork_15006615732289
// MI455X (gfx1250) — matrix operand order repaired and checked by reading, not compiled, not run on hardware
//
#include <hip/hip_runtime.h>
#include <hip/hip_bf16.h>

typedef __attribute__((ext_vector_type(16))) _Float16 v16h;
typedef __attribute__((ext_vector_type(8)))  _Float16 h8;
typedef __attribute__((ext_vector_type(8)))  float    v8f;

// ---------------------------------------------------------------------------
// Farthest point sampling: one block per batch, xyz + dists resident in LDS.
// ---------------------------------------------------------------------------
__global__ __launch_bounds__(256)
void fps_kernel(const float* __restrict__ xyz, int N, int npoint, int* __restrict__ outIdx)
{
    const int b = blockIdx.x;
    const float* p = xyz + (size_t)b * N * 3;
    __shared__ float px[1024], py[1024], pz[1024], dist[1024];
    __shared__ float rv[256];
    __shared__ int   ri[256];

    for (int j = threadIdx.x; j < N; j += blockDim.x) {
        px[j] = p[j * 3 + 0];
        py[j] = p[j * 3 + 1];
        pz[j] = p[j * 3 + 2];
        dist[j] = 1e10f;
    }
    if (threadIdx.x == 0) outIdx[(size_t)b * npoint] = 0;
    __syncthreads();

    int last = 0;
    for (int it = 1; it < npoint; ++it) {
        const float lx = px[last], ly = py[last], lz = pz[last];
        float bestV = -1.0f; int bestI = 0;
        for (int j = threadIdx.x; j < N; j += blockDim.x) {
            float dx = px[j] - lx, dy = py[j] - ly, dz = pz[j] - lz;
            float d  = dx * dx + dy * dy + dz * dz;
            float dm = fminf(dist[j], d);
            dist[j] = dm;
            if (dm > bestV) { bestV = dm; bestI = j; }   // strict >: keeps first index
        }
        rv[threadIdx.x] = bestV; ri[threadIdx.x] = bestI;
        __syncthreads();
        for (int st = 128; st > 0; st >>= 1) {
            if (threadIdx.x < st) {
                float v2 = rv[threadIdx.x + st]; int i2 = ri[threadIdx.x + st];
                float v1 = rv[threadIdx.x];      int i1 = ri[threadIdx.x];
                if (v2 > v1 || (v2 == v1 && i2 < i1)) { rv[threadIdx.x] = v2; ri[threadIdx.x] = i2; }
            }
            __syncthreads();
        }
        last = ri[0];
        if (threadIdx.x == 0) outIdx[(size_t)b * npoint + it] = last;
        __syncthreads();
    }
}

// ---------------------------------------------------------------------------
// Ball query: one thread per (b,s). Collect first nsample in-radius indices in
// ascending order; pad with first hit (or 0 if none). Also writes centers.
// ---------------------------------------------------------------------------
__global__ __launch_bounds__(256)
void ballq_kernel(const float* __restrict__ xyz, const int* __restrict__ cidx,
                  int N, int S, int nsample, float radius,
                  int* __restrict__ nidx, float* __restrict__ centers, int BS)
{
    int idx = blockIdx.x * blockDim.x + threadIdx.x;
    if (idx >= BS) return;
    int b = idx / S;
    const float* p = xyz + (size_t)b * N * 3;
    int ci = cidx[idx];
    float cx = p[ci * 3 + 0], cy = p[ci * 3 + 1], cz = p[ci * 3 + 2];
    centers[(size_t)idx * 3 + 0] = cx;
    centers[(size_t)idx * 3 + 1] = cy;
    centers[(size_t)idx * 3 + 2] = cz;

    int* out = nidx + (size_t)idx * nsample;
    const float r2 = radius * radius;
    int cnt = 0, first = 0;
    for (int j = 0; j < N; ++j) {
        float dx = p[j * 3 + 0] - cx, dy = p[j * 3 + 1] - cy, dz = p[j * 3 + 2] - cz;
        float d = dx * dx + dy * dy + dz * dz;
        if (d < r2) {
            if (cnt == 0) first = j;
            if (cnt < nsample) out[cnt] = j;
            ++cnt;
        }
    }
    int kept = cnt < nsample ? cnt : nsample;
    for (int k = kept; k < nsample; ++k) out[k] = first;
}

// ---------------------------------------------------------------------------
// Gather + relative coords -> f16 activation matrix X[Cpad][B*S*K]; channels
// [3+Cf .. Cpad) are zero so the GEMM needs no K-dim guards.
// ---------------------------------------------------------------------------
__global__ __launch_bounds__(256)
void group_kernel(const float* __restrict__ xyz, const float* __restrict__ feats,
                  const int* __restrict__ nidx, const float* __restrict__ centers,
                  _Float16* __restrict__ X, int N, int S, int K, int Cf, int Cpad,
                  long long Mll)
{
    size_t M = (size_t)Mll;
    size_t m = (size_t)blockIdx.x * blockDim.x + threadIdx.x;
    if (m >= M) return;
    int k = (int)(m % K);
    size_t bs = m / K;
    int b = (int)(bs / S);
    int j = nidx[bs * K + k];
    const float* pp = xyz + ((size_t)b * N + j) * 3;
    const float* cc = centers + bs * 3;
    X[0 * M + m] = (_Float16)(pp[0] - cc[0]);
    X[1 * M + m] = (_Float16)(pp[1] - cc[1]);
    X[2 * M + m] = (_Float16)(pp[2] - cc[2]);
    const float* f = feats + (size_t)b * Cf * N;
    for (int c = 0; c < Cf; ++c)
        X[(size_t)(3 + c) * M + m] = (_Float16)f[(size_t)c * N + j];
    for (int c = 3 + Cf; c < Cpad; ++c)
        X[(size_t)c * M + m] = (_Float16)0.0f;
}

// SA3 grouping: absolute xyz channels + feats, S=1, K points. m = b*K + k.
__global__ __launch_bounds__(256)
void group3_kernel(const float* __restrict__ xyz, const float* __restrict__ feats,
                   _Float16* __restrict__ X, int K, int Cf, int Cpad, int Mtot)
{
    int m = blockIdx.x * blockDim.x + threadIdx.x;
    if (m >= Mtot) return;
    size_t M = (size_t)Mtot;
    int k = m % K, b = m / K;
    const float* pp = xyz + ((size_t)b * K + k) * 3;
    X[0 * M + m] = (_Float16)pp[0];
    X[1 * M + m] = (_Float16)pp[1];
    X[2 * M + m] = (_Float16)pp[2];
    const float* f = feats + (size_t)b * Cf * K;
    for (int c = 0; c < Cf; ++c)
        X[(size_t)(3 + c) * M + m] = (_Float16)f[(size_t)c * K + k];
    for (int c = 3 + Cf; c < Cpad; ++c)
        X[(size_t)c * M + m] = (_Float16)0.0f;
}

// Weight pre-conversion: f32 [O][C] -> f16 zero-padded [O][Cpad] (16B aligned rows).
__global__ __launch_bounds__(256)
void wconv_kernel(const float* __restrict__ W, _Float16* __restrict__ Wh,
                  int C, int Cpad, int total)
{
    int i = blockIdx.x * blockDim.x + threadIdx.x;
    if (i >= total) return;
    int c = i % Cpad, o = i / Cpad;
    Wh[i] = (c < C) ? (_Float16)W[(size_t)o * C + c] : (_Float16)0.0f;
}

// ---------------------------------------------------------------------------
// WMMA GEMM: Y[O][M] = Wh[O][Cpad] * X[Cpad][M] (+bias). f16 in, f32 acc.
// Block = (blockDim/32) o-tiles x NT m-tiles. X panel staged TRANSPOSED in
// LDS (shT[col][k], padded stride) so each lane's 16 B values are two 16-byte runs
// (k 8h..8h+7, 16+8h..16+8h+7) -> two ds_load_b128 per WMMA. A fragment = two contiguous
// global b128 loads, reused across NT accumulators. No K-loop conditionals.
// ---------------------------------------------------------------------------
template <int NT>
__global__ __launch_bounds__(256)
void wmma_gemm_f16(const _Float16* __restrict__ X, const _Float16* __restrict__ Wh,
                   const float* __restrict__ bias, _Float16* __restrict__ Y,
                   int Cpad, int M, int O)
{
    extern __shared__ __align__(16) _Float16 shT[];   // [16*NT][Cpad+8]
    const int strideH = Cpad + 8;
    constexpr int NC  = 16 * NT;
    constexpr int SHC = (NT == 2) ? 5 : 4;
    const int m0 = blockIdx.x * NC;

    // cooperative transposed stage of X[:, m0:m0+NC]
    const int panel = Cpad << SHC;
    for (int idx = threadIdx.x; idx < panel; idx += blockDim.x) {
        int col = idx & (NC - 1), row = idx >> SHC;
        shT[col * strideH + row] = X[(size_t)row * M + (m0 + col)];
    }
    __syncthreads();

    const int wid  = threadIdx.x >> 5;
    const int lane = threadIdx.x & 31;
    const int o0   = (blockIdx.y * (blockDim.x >> 5) + wid) << 4;

    const int  r   = lane & 15;
    const bool hi  = lane >= 16;
    const int  kbA = hi ? 8 : 0;   // A: lanes 0-15 K 0-7/16-23, lanes 16-31 K 8-15/24-31
    const int  kbB = hi ? 8 : 0;   // B: lanes 0-15 K 0-7/16-23, lanes 16-31 K 8-15/24-31

    const _Float16* wrow = Wh + (size_t)(o0 + r) * Cpad;
    v8f acc[NT];
#pragma unroll
    for (int t = 0; t < NT; ++t) acc[t] = (v8f){};

    for (int k0 = 0; k0 < Cpad; k0 += 32) {
        h8 a0 = *(const h8*)(wrow + k0 + kbA);         // 16B aligned contiguous
        h8 a1 = *(const h8*)(wrow + k0 + kbA + 16);
        __builtin_prefetch(wrow + k0 + 32, 0, 3);
        v16h a;
#pragma unroll
        for (int t = 0; t < 8; ++t) { a[t] = a0[t]; a[8 + t] = a1[t]; }

#pragma unroll
        for (int t = 0; t < NT; ++t) {
            const h8* bp = (const h8*)(shT + (t * 16 + r) * strideH + k0 + kbB);
            h8 b0 = bp[0], b1 = bp[2];                 // 2x ds_load_b128
            v16h bb;
#pragma unroll
            for (int j = 0; j < 8; ++j) { bb[j] = b0[j]; bb[8 + j] = b1[j]; }
            acc[t] = __builtin_amdgcn_wmma_f32_16x16x32_f16(false, a, false, bb,
                                                            (short)0, acc[t], false, false);
        }
    }

    const int rowOff = hi ? 8 : 0;
    const int col    = lane & 15;
#pragma unroll
    for (int t = 0; t < NT; ++t) {
#pragma unroll
        for (int v = 0; v < 8; ++v) {
            int orow = o0 + rowOff + v;
            float val = acc[t][v];
            if (bias) val += bias[orow];
            Y[(size_t)orow * M + (m0 + t * 16 + col)] = (_Float16)val;
        }
    }
}

// ---------------------------------------------------------------------------
// BN stats: one block per output channel, reduce M elements -> scale/shift.
// ---------------------------------------------------------------------------
__global__ __launch_bounds__(256)
void bn_stats_kernel(const _Float16* __restrict__ Y, int M,
                     const float* __restrict__ gamma, const float* __restrict__ beta,
                     float* __restrict__ scale, float* __restrict__ shift)
{
    const int o = blockIdx.x;
    const _Float16* row = Y + (size_t)o * M;
    float s = 0.0f, sq = 0.0f;
    for (int i = threadIdx.x; i < M; i += blockDim.x) {
        float v = (float)row[i];
        s += v; sq += v * v;
    }
    __shared__ float ss[256], ssq[256];
    ss[threadIdx.x] = s; ssq[threadIdx.x] = sq;
    __syncthreads();
    for (int st = 128; st > 0; st >>= 1) {
        if (threadIdx.x < st) {
            ss[threadIdx.x]  += ss[threadIdx.x + st];
            ssq[threadIdx.x] += ssq[threadIdx.x + st];
        }
        __syncthreads();
    }
    if (threadIdx.x == 0) {
        float mean = ss[0] / (float)M;
        float var  = ssq[0] / (float)M - mean * mean;
        var = var < 0.0f ? 0.0f : var;
        float a = gamma[o] * rsqrtf(var + 1e-5f);
        scale[o] = a;
        shift[o] = beta[o] - a * mean;
    }
}

// 2D grid: blockIdx.y = channel; no per-element division.
__global__ __launch_bounds__(256)
void bn_apply_relu_kernel(_Float16* __restrict__ Y, const float* __restrict__ scale,
                          const float* __restrict__ shift, int M)
{
    int o = blockIdx.y;
    int i = blockIdx.x * blockDim.x + threadIdx.x;
    if (i >= M) return;
    size_t idx = (size_t)o * M + i;
    float v = (float)Y[idx] * scale[o] + shift[o];
    Y[idx] = (_Float16)(v > 0.0f ? v : 0.0f);
}

// Max over K -> f32 [b][O][S]
__global__ __launch_bounds__(256)
void maxpool_kernel(const _Float16* __restrict__ Y, float* __restrict__ out,
                    int B, int O, int S, int K)
{
    int idx = blockIdx.x * blockDim.x + threadIdx.x;
    if (idx >= B * O * S) return;
    int s = idx % S; int t = idx / S; int o = t % O; int b = t / O;
    size_t M = (size_t)B * S * K;
    const _Float16* row = Y + (size_t)o * M + ((size_t)b * S + s) * K;
    float m = -3.0e38f;
    for (int k = 0; k < K; ++k) m = fmaxf(m, (float)row[k]);
    out[((size_t)b * O + o) * S + s] = m;
}

// Max over K -> f16 transposed [O][B]  (feeds FC GEMM directly)
__global__ __launch_bounds__(256)
void maxpool_t_kernel(const _Float16* __restrict__ Y, _Float16* __restrict__ out,
                      int B, int O, int K)
{
    int idx = blockIdx.x * blockDim.x + threadIdx.x;
    if (idx >= B * O) return;
    int b = idx % B; int o = idx / B;
    size_t M = (size_t)B * K;
    const _Float16* row = Y + (size_t)o * M + (size_t)b * K;
    float m = -3.0e38f;
    for (int k = 0; k < K; ++k) m = fmaxf(m, (float)row[k]);
    out[(size_t)o * B + b] = (_Float16)m;
}

// Head: out[b] = sum_o X[o][b]*Wh[o] + bh
__global__ __launch_bounds__(256)
void head_kernel(const _Float16* __restrict__ X, const float* __restrict__ Wh,
                 const float* __restrict__ bh, float* __restrict__ out, int H, int B)
{
    int b = blockIdx.x;
    float s = 0.0f;
    for (int o = threadIdx.x; o < H; o += blockDim.x)
        s += (float)X[(size_t)o * B + b] * Wh[o];
    __shared__ float ss[256];
    ss[threadIdx.x] = s;
    __syncthreads();
    for (int st = 128; st > 0; st >>= 1) {
        if (threadIdx.x < st) ss[threadIdx.x] += ss[threadIdx.x + st];
        __syncthreads();
    }
    if (threadIdx.x == 0) out[b] = ss[0] + bh[0];
}

// ---------------------------------------------------------------------------
extern "C" void kernel_launch(void* const* d_in, const int* in_sizes, int n_in,
                              void* d_out, int out_size, void* d_ws, size_t ws_size,
                              hipStream_t stream)
{
    (void)in_sizes; (void)n_in; (void)out_size; (void)ws_size;
    const int B = 64, N = 1024;

    const float* pc = (const float*)d_in[0];
    const float* pf = (const float*)d_in[1];

    const float* saW[3][3]; const float* saG[3][3]; const float* saB[3][3];
    int ii = 2;
    for (int s = 0; s < 3; ++s)
        for (int l = 0; l < 3; ++l) {
            saW[s][l] = (const float*)d_in[ii++];
            saG[s][l] = (const float*)d_in[ii++];
            saB[s][l] = (const float*)d_in[ii++];
        }
    const float* fcW[2]; const float* fcB[2]; const float* fcG[2]; const float* fcBe[2];
    for (int l = 0; l < 2; ++l) {
        fcW[l]  = (const float*)d_in[ii++];
        fcB[l]  = (const float*)d_in[ii++];
        fcG[l]  = (const float*)d_in[ii++];
        fcBe[l] = (const float*)d_in[ii++];
    }
    const float* headW = (const float*)d_in[ii++];
    const float* headB = (const float*)d_in[ii++];

    // ---- workspace carve-up --------------------------------------------
    char* ws = (char*)d_ws;
    size_t off = 0;
    auto take = [&](size_t bytes) -> char* {
        char* p = ws + off;
        off = (off + bytes + 255) & ~(size_t)255;
        return p;
    };
    int*      cidx1    = (int*)     take((size_t)B * 128 * 4);
    float*    centers1 = (float*)   take((size_t)B * 128 * 3 * 4);
    int*      nidx1    = (int*)     take((size_t)B * 128 * 64 * 4);
    int*      cidx2    = (int*)     take((size_t)B * 32 * 4);
    float*    centers2 = (float*)   take((size_t)B * 32 * 3 * 4);
    int*      nidx2    = (int*)     take((size_t)B * 32 * 128 * 4);
    float*    feats1   = (float*)   take((size_t)B * 128 * 128 * 4);
    float*    feats2   = (float*)   take((size_t)B * 256 * 32 * 4);
    float*    bnScale  = (float*)   take(1024 * 4);
    float*    bnShift  = (float*)   take(1024 * 4);
    _Float16* whBuf    = (_Float16*)take((size_t)1024 * 1024 * 2);    // padded f16 weights
    _Float16* bufA     = (_Float16*)take((size_t)160 * 262144 * 2);   // ~84 MB (X panels)
    _Float16* bufB     = (_Float16*)take((size_t)128 * 524288 * 2);   // ~134 MB

    auto gemmBN = [&](const _Float16* X, const float* W, const float* bias, _Float16* Y,
                      int C, int M, int O, const float* g, const float* bt) {
        int Cpad = (C + 31) & ~31;
        int wct  = O * Cpad;
        wconv_kernel<<<(wct + 255) / 256, 256, 0, stream>>>(W, whBuf, C, Cpad, wct);
        int TO = O / 16; if (TO > 8) TO = 8;
        int NT = (Cpad <= 512) ? 2 : 1;
        size_t shmem = (size_t)16 * NT * (Cpad + 8) * 2;
        dim3 grid(M / (16 * NT), O / (16 * TO));
        if (NT == 2)
            wmma_gemm_f16<2><<<grid, 32 * TO, shmem, stream>>>(X, whBuf, bias, Y, Cpad, M, O);
        else
            wmma_gemm_f16<1><<<grid, 32 * TO, shmem, stream>>>(X, whBuf, bias, Y, Cpad, M, O);
        bn_stats_kernel<<<O, 256, 0, stream>>>(Y, M, g, bt, bnScale, bnShift);
        bn_apply_relu_kernel<<<dim3((M + 255) / 256, O), 256, 0, stream>>>(Y, bnScale, bnShift, M);
    };

    // ================= SA1: N=1024 -> S=128, K=64, ch 7->64->64->128 =====
    {
        const int S = 128, K = 64;
        const long long M1 = (long long)B * S * K;          // 524288
        fps_kernel<<<B, 256, 0, stream>>>(pc, N, S, cidx1);
        ballq_kernel<<<(B * S + 255) / 256, 256, 0, stream>>>(pc, cidx1, N, S, K, 0.02f,
                                                              nidx1, centers1, B * S);
        group_kernel<<<(int)((M1 + 255) / 256), 256, 0, stream>>>(pc, pf, nidx1, centers1,
                                                                  bufA, N, S, K, 4, 32, M1);
        gemmBN(bufA, saW[0][0], nullptr, bufB, 7,  (int)M1, 64,  saG[0][0], saB[0][0]);
        gemmBN(bufB, saW[0][1], nullptr, bufA, 64, (int)M1, 64,  saG[0][1], saB[0][1]);
        gemmBN(bufA, saW[0][2], nullptr, bufB, 64, (int)M1, 128, saG[0][2], saB[0][2]);
        maxpool_kernel<<<(B * 128 * S + 255) / 256, 256, 0, stream>>>(bufB, feats1, B, 128, S, K);
    }

    // ================= SA2: N=128 -> S=32, K=128, ch 131->128->128->256 ==
    {
        const int Np = 128, S = 32, K = 128;
        const long long M2 = (long long)B * S * K;          // 262144
        fps_kernel<<<B, 256, 0, stream>>>(centers1, Np, S, cidx2);
        ballq_kernel<<<(B * S + 255) / 256, 256, 0, stream>>>(centers1, cidx2, Np, S, K, 0.04f,
                                                              nidx2, centers2, B * S);
        group_kernel<<<(int)((M2 + 255) / 256), 256, 0, stream>>>(centers1, feats1, nidx2,
                                                                  centers2, bufA, Np, S, K,
                                                                  128, 160, M2);
        gemmBN(bufA, saW[1][0], nullptr, bufB, 131, (int)M2, 128, saG[1][0], saB[1][0]);
        gemmBN(bufB, saW[1][1], nullptr, bufA, 128, (int)M2, 128, saG[1][1], saB[1][1]);
        gemmBN(bufA, saW[1][2], nullptr, bufB, 128, (int)M2, 256, saG[1][2], saB[1][2]);
        maxpool_kernel<<<(B * 256 * S + 255) / 256, 256, 0, stream>>>(bufB, feats2, B, 256, S, K);
    }

    // ================= SA3: global, K=32, ch 259->256->256->512 ==========
    {
        const int K = 32;
        const int M3 = B * K;                                // 2048
        group3_kernel<<<(M3 + 255) / 256, 256, 0, stream>>>(centers2, feats2, bufA, K, 256, 288, M3);
        gemmBN(bufA, saW[2][0], nullptr, bufB, 259, M3, 256, saG[2][0], saB[2][0]);
        gemmBN(bufB, saW[2][1], nullptr, bufA, 256, M3, 256, saG[2][1], saB[2][1]);
        gemmBN(bufA, saW[2][2], nullptr, bufB, 256, M3, 512, saG[2][2], saB[2][2]);
        // pooled feats (B,512) transposed -> Xfc [512][64] in bufA
        maxpool_t_kernel<<<(512 * B + 255) / 256, 256, 0, stream>>>(bufB, bufA, B, 512, K);
    }

    // ================= FC head: 512->1024->1024->1 =======================
    gemmBN(bufA, fcW[0], fcB[0], bufB, 512,  B, 1024, fcG[0], fcBe[0]);
    gemmBN(bufB, fcW[1], fcB[1], bufA, 1024, B, 1024, fcG[1], fcBe[1]);
    head_kernel<<<B, 256, 0, stream>>>(bufA, headW, headB, (float*)d_out, 1024, B);
}
